// RSNN_2l_15668040696087
// MI455X (gfx1250) — matrix operand order repaired and checked by reading, not compiled, not run on hardware
//
#include <hip/hip_runtime.h>
#include <hip/hip_bf16.h>

typedef __attribute__((ext_vector_type(16))) __bf16 v16bf;
typedef __attribute__((ext_vector_type(8)))  __bf16 v8bf;
typedef __attribute__((ext_vector_type(8)))  float  v8f;

#define B_TOTAL 256
#define T_STEPS 50
#define I_DIM   700
#define I_PAD   704
#define H_DIM   2048
#define O_DIM   20
#define O_PAD   32
#define M_TILE  16
#define THREADS 1024       // 32 waves: fills one WGP, minimizes per-lane persistent state
#define N_TILES 4          // N-tiles (16 cols) per wave: 32 waves * 64 cols = 2048
#define SPIKE_THRESH 0.3f

__device__ __forceinline__ float sigmoidf_(float v) { return 1.0f / (1.0f + __expf(-v)); }

// Accumulate one 16x16 f32 tile of  act[16xK] @ W^T[Kx16]  via bf16 WMMA.
// arow : LDS row pointer for this lane's M row (A layout: lane m=L%16, K split by half)
// wrow : global row pointer of weight row n (B layout: lane n=L%16, K split by half)
__device__ __forceinline__ v8f gemm_rowtile(v8f acc, const __bf16* arow, const __bf16* wrow,
                                            int K, int half) {
  #pragma unroll 2
  for (int kc = 0; kc < K; kc += 32) {
    v16bf a = *(const v16bf*)(arow + kc + half * 16);
    v16bf b = *(const v16bf*)(wrow + kc + half * 16);
    acc = __builtin_amdgcn_wmma_f32_16x16x32_bf16(false, a, false, b, (short)0, acc,
                                                  false, false);
  }
  return acc;
}

// fp32 -> bf16 with zero padding to (drows, dcols)
__global__ void cvt_bf16_pad(const float* __restrict__ src, __bf16* __restrict__ dst,
                             int srows, int scols, int drows, int dcols) {
  int n = drows * dcols;
  for (int i = blockIdx.x * blockDim.x + threadIdx.x; i < n; i += gridDim.x * blockDim.x) {
    int r = i / dcols, c = i - r * dcols;
    float v = (r < srows && c < scols) ? src[(size_t)r * scols + c] : 0.0f;
    dst[i] = (__bf16)v;
  }
}

__global__ __launch_bounds__(THREADS, 1)
void rsnn_kernel(const float* __restrict__ x,
                 const float* __restrict__ tau1,
                 const float* __restrict__ tau2,
                 const float* __restrict__ tauo,
                 const __bf16* __restrict__ Wih,
                 const __bf16* __restrict__ W11,
                 const __bf16* __restrict__ W12,
                 const __bf16* __restrict__ W22,
                 const __bf16* __restrict__ Who,
                 float* __restrict__ out)
{
  __shared__ alignas(32) __bf16 x_bf[M_TILE * I_PAD];   // staged x_t (bf16, padded)
  __shared__ alignas(32) __bf16 s1[M_TILE * H_DIM];     // layer-1 spikes
  __shared__ alignas(32) __bf16 s2[M_TILE * H_DIM];     // layer-2 spikes
  __shared__ float og[M_TILE * O_PAD];                  // output-layer GEMM result
  __shared__ float om_s[M_TILE * O_DIM];
  __shared__ float osp_s[M_TILE * O_DIM];
  __shared__ float osum_s[M_TILE * O_DIM];
  __shared__ float omot_s[M_TILE * O_DIM];
  __shared__ float ao_s[O_DIM];

  const int tid  = threadIdx.x;
  const int wave = tid >> 5;        // gfx1250 is wave32
  const int lane = tid & 31;
  const int lcol = lane & 15;
  const int half = lane >> 4;
  const int b0   = blockIdx.x * M_TILE;

  // ---- init shared state ----
  for (int i = tid; i < M_TILE * H_DIM; i += THREADS) {
    s1[i] = (__bf16)0.0f;
    s2[i] = (__bf16)0.0f;
  }
  if (tid < M_TILE * O_DIM) {
    om_s[tid] = 0.f; osp_s[tid] = 0.f; osum_s[tid] = 0.f; omot_s[tid] = 0.f;
  }
  if (tid < O_DIM) ao_s[tid] = sigmoidf_(tauo[tid]);

  // per-lane leak factors for the columns this lane owns (C/D layout: col = n0 + lane%16)
  float a1v[N_TILES], a2v[N_TILES];
  #pragma unroll
  for (int j = 0; j < N_TILES; ++j) {
    int col = wave * (N_TILES * 16) + j * 16 + lcol;
    a1v[j] = sigmoidf_(tau1[col]);
    a2v[j] = sigmoidf_(tau2[col]);
  }

  // persistent membrane potentials in registers (C/D tile layout): 64 VGPRs/lane
  v8f h1m[N_TILES], h2m[N_TILES];
  #pragma unroll
  for (int j = 0; j < N_TILES; ++j)
    #pragma unroll
    for (int r = 0; r < 8; ++r) { h1m[j][r] = 0.f; h2m[j][r] = 0.f; }

  __syncthreads();

  for (int t = 0; t < T_STEPS; ++t) {
    // ---- stage x_t -> bf16 LDS ----
    for (int i = tid; i < M_TILE * I_PAD; i += THREADS) {
      int row = i / I_PAD, col = i - row * I_PAD;
      float v = (col < I_DIM) ? x[((size_t)(b0 + row) * T_STEPS + t) * I_DIM + col] : 0.f;
      x_bf[i] = (__bf16)v;
    }
    __syncthreads();

    // ---- layer 1: h1m = x@Wih^T + s1@W11^T + h1m*a1*(1-s1) ----
    #pragma unroll
    for (int j = 0; j < N_TILES; ++j) {
      int col = wave * (N_TILES * 16) + j * 16 + lcol;
      v8f acc;
      #pragma unroll
      for (int r = 0; r < 8; ++r) {
        int m = r + 8 * half;
        float sp = (float)s1[m * H_DIM + col];
        acc[r] = h1m[j][r] * a1v[j] * (1.0f - sp);
      }
      acc = gemm_rowtile(acc, x_bf + lcol * I_PAD, Wih + (size_t)col * I_PAD, I_PAD, half);
      acc = gemm_rowtile(acc, s1  + lcol * H_DIM, W11 + (size_t)col * H_DIM, H_DIM, half);
      h1m[j] = acc;
    }
    __syncthreads();                 // all reads of old s1 complete
    #pragma unroll
    for (int j = 0; j < N_TILES; ++j) {
      int col = wave * (N_TILES * 16) + j * 16 + lcol;
      #pragma unroll
      for (int r = 0; r < 8; ++r)
        s1[(r + 8 * half) * H_DIM + col] = (__bf16)((h1m[j][r] > SPIKE_THRESH) ? 1.0f : 0.0f);
    }
    __syncthreads();                 // new s1 visible

    // ---- layer 2: h2m = s1@W12^T + s2@W22^T + h2m*a2*(1-s2) ----
    #pragma unroll
    for (int j = 0; j < N_TILES; ++j) {
      int col = wave * (N_TILES * 16) + j * 16 + lcol;
      v8f acc;
      #pragma unroll
      for (int r = 0; r < 8; ++r) {
        int m = r + 8 * half;
        float sp = (float)s2[m * H_DIM + col];
        acc[r] = h2m[j][r] * a2v[j] * (1.0f - sp);
      }
      acc = gemm_rowtile(acc, s1 + lcol * H_DIM, W12 + (size_t)col * H_DIM, H_DIM, half);
      acc = gemm_rowtile(acc, s2 + lcol * H_DIM, W22 + (size_t)col * H_DIM, H_DIM, half);
      h2m[j] = acc;
    }
    __syncthreads();
    #pragma unroll
    for (int j = 0; j < N_TILES; ++j) {
      int col = wave * (N_TILES * 16) + j * 16 + lcol;
      #pragma unroll
      for (int r = 0; r < 8; ++r)
        s2[(r + 8 * half) * H_DIM + col] = (__bf16)((h2m[j][r] > SPIKE_THRESH) ? 1.0f : 0.0f);
    }
    __syncthreads();

    // ---- output-layer GEMM: og = s2 @ Who^T  (16 x 32, padded) ----
    if (wave < 2) {                  // uniform per-wave branch: EXEC all-1 for WMMA
      int col = wave * 16 + lcol;
      v8f acc;
      #pragma unroll
      for (int r = 0; r < 8; ++r) acc[r] = 0.f;
      acc = gemm_rowtile(acc, s2 + lcol * H_DIM, Who + (size_t)col * H_DIM, H_DIM, half);
      #pragma unroll
      for (int r = 0; r < 8; ++r) og[(r + 8 * half) * O_PAD + col] = acc[r];
    }
    __syncthreads();

    // ---- output recurrence + spike/softmax accumulation, one batch row per lane ----
    if (tid < M_TILE) {
      float omv[O_DIM];
      #pragma unroll
      for (int o = 0; o < O_DIM; ++o) {
        float v = om_s[tid * O_DIM + o] * ao_s[o] * (1.0f - osp_s[tid * O_DIM + o])
                + og[tid * O_PAD + o];
        om_s[tid * O_DIM + o] = v;
        omv[o] = v;
        float sp = (v > SPIKE_THRESH) ? 1.0f : 0.0f;
        osp_s[tid * O_DIM + o] = sp;
        osum_s[tid * O_DIM + o] += sp;
      }
      float mx = omv[0];
      #pragma unroll
      for (int o = 1; o < O_DIM; ++o) mx = fmaxf(mx, omv[o]);
      float ex[O_DIM];
      float se = 0.f;
      #pragma unroll
      for (int o = 0; o < O_DIM; ++o) { ex[o] = __expf(omv[o] - mx); se += ex[o]; }
      float inv = 1.0f / se;
      #pragma unroll
      for (int o = 0; o < O_DIM; ++o) omot_s[tid * O_DIM + o] += ex[o] * inv;
    }
    __syncthreads();
  }

  // ---- final write: [osum/WIN ; omot], each [256,20] flat, concatenated ----
  if (tid < M_TILE) {
    #pragma unroll
    for (int o = 0; o < O_DIM; ++o) {
      out[(size_t)(b0 + tid) * O_DIM + o] = osum_s[tid * O_DIM + o] * (1.0f / (float)T_STEPS);
      out[(size_t)B_TOTAL * O_DIM + (size_t)(b0 + tid) * O_DIM + o] = omot_s[tid * O_DIM + o];
    }
  }
}

extern "C" void kernel_launch(void* const* d_in, const int* in_sizes, int n_in,
                              void* d_out, int out_size, void* d_ws, size_t ws_size,
                              hipStream_t stream) {
  (void)in_sizes; (void)n_in; (void)out_size; (void)ws_size;
  const float* x    = (const float*)d_in[0];
  const float* wih  = (const float*)d_in[1];
  const float* w11  = (const float*)d_in[2];
  const float* w12  = (const float*)d_in[3];
  const float* w22  = (const float*)d_in[4];
  const float* who  = (const float*)d_in[5];
  const float* tau1 = (const float*)d_in[6];
  const float* tau2 = (const float*)d_in[7];
  const float* tauo = (const float*)d_in[8];

  // workspace layout: bf16 weight copies (padded to WMMA tile multiples), ~28.2 MB
  char* ws = (char*)d_ws;
  size_t off = 0;
  __bf16* Wih = (__bf16*)(ws + off); off += (size_t)H_DIM * I_PAD * 2;
  __bf16* W11 = (__bf16*)(ws + off); off += (size_t)H_DIM * H_DIM * 2;
  __bf16* W12 = (__bf16*)(ws + off); off += (size_t)H_DIM * H_DIM * 2;
  __bf16* W22 = (__bf16*)(ws + off); off += (size_t)H_DIM * H_DIM * 2;
  __bf16* Who = (__bf16*)(ws + off); off += (size_t)O_PAD * H_DIM * 2;

  auto cvt = [&](const float* src, __bf16* dst, int sr, int sc, int dr, int dc) {
    int n = dr * dc;
    int blocks = (n + 255) / 256;
    cvt_bf16_pad<<<blocks, 256, 0, stream>>>(src, dst, sr, sc, dr, dc);
  };
  cvt(wih, Wih, H_DIM, I_DIM, H_DIM, I_PAD);
  cvt(w11, W11, H_DIM, H_DIM, H_DIM, H_DIM);
  cvt(w12, W12, H_DIM, H_DIM, H_DIM, H_DIM);
  cvt(w22, W22, H_DIM, H_DIM, H_DIM, H_DIM);
  cvt(who, Who, O_DIM, H_DIM, O_PAD, H_DIM);

  rsnn_kernel<<<B_TOTAL / M_TILE, THREADS, 0, stream>>>(
      x, tau1, tau2, tauo, Wih, W11, W12, W22, Who, (float*)d_out);
}
